// MambaVQVAE_14096082665951
// MI455X (gfx1250) — compile-verified, matrix operand order checked by reading, not run on hardware
//
#include <hip/hip_runtime.h>
#include <hip/hip_bf16.h>
#include <stdint.h>

typedef __attribute__((ext_vector_type(16))) __bf16   v16bf;
typedef __attribute__((ext_vector_type(8)))  float    v8f;
typedef __attribute__((ext_vector_type(8)))  uint32_t v8u;

#define NROWS   73728      // 128*24*1536 / 64
#define NCODES  4096
#define DIM     64
#define NKT     256        // NCODES / 16 code tiles
#define ZELEMS  4718592    // NROWS * DIM

// ---- d_out flat offsets (tuple order: loss, z_q_st, codes, new_emb, new_cs, new_ema_w)
#define OFF_LOSS 0
#define OFF_ZQ   1
#define OFF_CODE (1 + ZELEMS)                  // 4718593
#define OFF_EMB  (OFF_CODE + NROWS)            // 4792321
#define OFF_NCS  (OFF_EMB + NCODES * DIM)      // 5054465
#define OFF_NEW  (OFF_NCS + NCODES)            // 5058561

// ---- workspace dword offsets
#define WS_BHI   0            // packed bf16-hi B images: [256][2][32][8] u32 = 131072
#define WS_BLO   131072       // packed bf16-lo B images
#define WS_ENORM 262144       // ||e_k||^2, 4096 f32
#define WS_CNT   266240       // counts, 4096 f32 (zeroed)
#define WS_DW    270336       // dw, 4096*64 f32 (zeroed)
#define WS_CODE  532480       // codes, 73728 i32
#define WS_NPART 606208       // 16 partial sums of new_cluster_size
#define WS_LPART 606224       // 18432 per-block loss partials
#define ZERO_CNT (WS_CODE - WS_CNT)   // zero [WS_CNT, WS_CODE)

#define DECAY    0.99f
#define ONE_MD   0.01f
#define EPS      1e-5f
#define CCOST    0.25f

__device__ __forceinline__ void split_bf16(float x, __bf16 &h, __bf16 &l) {
    h = (__bf16)x;
    l = (__bf16)(x - (float)h);
}

// -------------------- scratch zero --------------------
__global__ __launch_bounds__(256) void k_zero(float* ws) {
    int i = blockIdx.x * 256 + threadIdx.x;
    if (i < ZERO_CNT) ws[WS_CNT + i] = 0.0f;
}

// -------------------- ||e_k||^2 --------------------
__global__ __launch_bounds__(256) void k_enorm(const float* __restrict__ emb, float* __restrict__ enorm) {
    int k = blockIdx.x * 256 + threadIdx.x;
    if (k >= NCODES) return;
    const float* e = emb + (size_t)k * DIM;
    float s = 0.f;
#pragma unroll
    for (int d = 0; d < DIM; ++d) s = fmaf(e[d], e[d], s);
    enorm[k] = s;
}

// -------------------- pack codebook into WMMA B-operand per-lane images --------------------
// B tile for code-tile T, k-chunk kc is the 32x16 (KxN) operand; stored as the per-lane
// register image: lane n(0..15) holds code T*16+n, dims {kc*32 + g*16 + 0..7}; lanes 16..31
// same codes with +8 dim offset (mirrors the documented 16-bit A layout for B^T).
__global__ __launch_bounds__(256) void k_packB(const float* __restrict__ emb,
                                               uint32_t* __restrict__ bhi,
                                               uint32_t* __restrict__ blo) {
    int gid = blockIdx.x * 256 + threadIdx.x;   // 256*2*32 = 16384 total
    if (gid >= NKT * 2 * 32) return;
    int lane = gid & 31;
    int kc   = (gid >> 5) & 1;
    int T    = gid >> 6;
    int n    = lane & 15;
    int hw   = lane >> 4;
    const float* er = emb + (size_t)(T * 16 + n) * DIM;
    uint32_t oh[8], ol[8];
#pragma unroll
    for (int d = 0; d < 8; ++d) { oh[d] = 0u; ol[d] = 0u; }
#pragma unroll
    for (int g = 0; g < 2; ++g) {
        int kbase = kc * 32 + g * 16 + hw * 8;
#pragma unroll
        for (int i = 0; i < 8; ++i) {
            __bf16 h, l;
            split_bf16(er[kbase + i], h, l);
            int j = g * 8 + i;
            int dw = j >> 1, sh = (j & 1) * 16;
            oh[dw] |= ((uint32_t)__builtin_bit_cast(unsigned short, h)) << sh;
            ol[dw] |= ((uint32_t)__builtin_bit_cast(unsigned short, l)) << sh;
        }
    }
    size_t base = ((((size_t)T * 2 + kc) * 32) + lane) * 8;
#pragma unroll
    for (int d = 0; d < 8; ++d) { bhi[base + d] = oh[d]; blo[base + d] = ol[d]; }
}

// -------------------- fused distance GEMM + argmin (WMMA bf16x3) --------------------
// 4 waves per block; each wave owns 32 rows (two 16-row M-tiles) -> block covers 128 rows.
// B-operand images are double-buffered in LDS via CDNA5 async global->LDS DMA
// (global_load_async_to_lds_b128, ASYNCcnt), overlapping tile T+1 staging with tile T WMMAs.
// LDS per-lane row stride is 12 dwords (48B): 16B aligned for ds_load_b128, low bank conflict.
__global__ __launch_bounds__(128) void k_argmin(const float* __restrict__ z,
                                                const uint32_t* __restrict__ bhi,
                                                const uint32_t* __restrict__ blo,
                                                const float* __restrict__ enorm,
                                                int*  __restrict__ codes_i,
                                                float* __restrict__ codes_f) {
    __shared__ __attribute__((aligned(16))) uint32_t lbuf[2][2][2][32][12]; // [buf][hilo][kc][lane][12]

    const int tid  = threadIdx.x;
    const int lane = tid & 31;
    const int wv   = tid >> 5;       // wave in block
    const int hw   = lane >> 4;      // half of wave
    const int m    = lane & 15;
    const int rowbase = blockIdx.x * 128 + wv * 32;

    // async-stage one code tile's B images (4KB) into LDS buffer b: 128 threads x 32B
    auto stage = [&](int T, int b) {
        int hilo = tid >> 6;
        int kc   = (tid >> 5) & 1;
        int ln   = tid & 31;
        const uint32_t* src = (hilo ? blo : bhi) + (((size_t)T * 2 + kc) * 32 + ln) * 8;
        uint32_t daddr = (uint32_t)(uintptr_t)&lbuf[b][hilo][kc][ln][0];
        asm volatile("global_load_async_to_lds_b128 %0, %1, off"
                     :: "v"(daddr), "v"(src) : "memory");
        asm volatile("global_load_async_to_lds_b128 %0, %1, off offset:16"
                     :: "v"(daddr), "v"(src) : "memory");
    };

    // Load + split A (z rows) into the 16-bit A-matrix per-lane layout.
    v16bf a_hi[2][2], a_lo[2][2];
#pragma unroll
    for (int t = 0; t < 2; ++t) {
        const float* zr = z + (size_t)(rowbase + t * 16 + m) * DIM;
#pragma unroll
        for (int kc = 0; kc < 2; ++kc) {
#pragma unroll
            for (int g = 0; g < 2; ++g) {
                int kbase = kc * 32 + g * 16 + hw * 8;
#pragma unroll
                for (int i = 0; i < 8; ++i) {
                    __bf16 h, l;
                    split_bf16(zr[kbase + i], h, l);
                    a_hi[t][kc][g * 8 + i] = h;
                    a_lo[t][kc][g * 8 + i] = l;
                }
            }
        }
    }

    float best[2][8];
    int   bidx[2][8];
#pragma unroll
    for (int t = 0; t < 2; ++t)
#pragma unroll
        for (int v = 0; v < 8; ++v) { best[t][v] = 3.402823466e+38f; bidx[t][v] = 0; }

    stage(0, 0);   // prologue

    for (int T = 0; T < NKT; ++T) {
        int b = T & 1;
        asm volatile("s_wait_asynccnt 0x0" ::: "memory");   // own DMA writes landed
        __syncthreads();                                    // whole buffer b ready
        if (T + 1 < NKT) stage(T + 1, b ^ 1);               // overlap next tile's DMA

        v8f acc[2];
#pragma unroll
        for (int t = 0; t < 2; ++t)
#pragma unroll
            for (int i = 0; i < 8; ++i) acc[t][i] = 0.0f;

#pragma unroll
        for (int kc = 0; kc < 2; ++kc) {
            const uint4* ph = (const uint4*)&lbuf[b][0][kc][lane][0];
            const uint4* pl = (const uint4*)&lbuf[b][1][kc][lane][0];
            uint4 h0 = ph[0], h1 = ph[1];
            uint4 l0 = pl[0], l1 = pl[1];
            v8u uh, ul;
            uh[0] = h0.x; uh[1] = h0.y; uh[2] = h0.z; uh[3] = h0.w;
            uh[4] = h1.x; uh[5] = h1.y; uh[6] = h1.z; uh[7] = h1.w;
            ul[0] = l0.x; ul[1] = l0.y; ul[2] = l0.z; ul[3] = l0.w;
            ul[4] = l1.x; ul[5] = l1.y; ul[6] = l1.z; ul[7] = l1.w;
            v16bf bh = __builtin_bit_cast(v16bf, uh);
            v16bf bl = __builtin_bit_cast(v16bf, ul);
#pragma unroll
            for (int t = 0; t < 2; ++t) {
                acc[t] = __builtin_amdgcn_wmma_f32_16x16x32_bf16(false, a_hi[t][kc], false, bh,
                                                                (short)0, acc[t], false, false);
                acc[t] = __builtin_amdgcn_wmma_f32_16x16x32_bf16(false, a_hi[t][kc], false, bl,
                                                                (short)0, acc[t], false, false);
                acc[t] = __builtin_amdgcn_wmma_f32_16x16x32_bf16(false, a_lo[t][kc], false, bh,
                                                                (short)0, acc[t], false, false);
            }
        }

        int   c  = T * 16 + m;          // C layout: lane column = lane&15
        float en = enorm[c];
#pragma unroll
        for (int t = 0; t < 2; ++t)
#pragma unroll
            for (int v = 0; v < 8; ++v) {
                float s = fmaf(-2.0f, acc[t][v], en);   // ||e||^2 - 2 z.e
                if (s < best[t][v]) { best[t][v] = s; bidx[t][v] = c; }
            }
    }

    // Reduce across the 16 lanes of each half (xor of bits 0..3 stays in-half).
#pragma unroll
    for (int mask = 1; mask < 16; mask <<= 1) {
#pragma unroll
        for (int t = 0; t < 2; ++t)
#pragma unroll
            for (int v = 0; v < 8; ++v) {
                float os = __shfl_xor(best[t][v], mask, 32);
                int   oi = __shfl_xor(bidx[t][v], mask, 32);
                if (os < best[t][v] || (os == best[t][v] && oi < bidx[t][v])) {
                    best[t][v] = os; bidx[t][v] = oi;
                }
            }
    }
    if (m == 0) {            // lanes 0 and 16: rows v (+8 for upper half)
#pragma unroll
        for (int t = 0; t < 2; ++t)
#pragma unroll
            for (int v = 0; v < 8; ++v) {
                int r = rowbase + t * 16 + hw * 8 + v;
                codes_i[r] = bidx[t][v];
                codes_f[r] = (float)bidx[t][v];
            }
    }
}

// -------------------- scatter: counts + dw --------------------
__global__ __launch_bounds__(256) void k_scatter(const float* __restrict__ z,
                                                 const int* __restrict__ codes_i,
                                                 float* __restrict__ counts,
                                                 float* __restrict__ dw) {
    int gid = blockIdx.x * 256 + threadIdx.x;     // NROWS*64 threads
    int row = gid >> 6, d = gid & 63;
    if (row >= NROWS) return;
    int c = codes_i[row];
    atomicAdd(&dw[(size_t)c * DIM + d], z[gid]);
    if (d == 0) atomicAdd(&counts[c], 1.0f);
}

// -------------------- EMA update + partial sums of n --------------------
__global__ __launch_bounds__(256) void k_ema(const float* __restrict__ ecs,
                                             const float* __restrict__ emaw,
                                             const float* __restrict__ counts,
                                             const float* __restrict__ dw,
                                             float* __restrict__ out_ncs,
                                             float* __restrict__ out_nemaw,
                                             float* __restrict__ npart) {
    int k = blockIdx.x * 256 + threadIdx.x;       // 4096 threads, 16 blocks
    float ncs = DECAY * ecs[k] + ONE_MD * counts[k];
    out_ncs[k] = ncs;
    size_t base = (size_t)k * DIM;
#pragma unroll 8
    for (int d = 0; d < DIM; ++d)
        out_nemaw[base + d] = DECAY * emaw[base + d] + ONE_MD * dw[base + d];
    __shared__ float sm[256];
    sm[threadIdx.x] = ncs;
    __syncthreads();
    for (int s = 128; s > 0; s >>= 1) {
        if (threadIdx.x < s) sm[threadIdx.x] += sm[threadIdx.x + s];
        __syncthreads();
    }
    if (threadIdx.x == 0) npart[blockIdx.x] = sm[0];   // deterministic: per-block slot
}

// -------------------- normalize: new embedding --------------------
__global__ __launch_bounds__(256) void k_newemb(const float* __restrict__ out_ncs,
                                                const float* __restrict__ out_nemaw,
                                                const float* __restrict__ npart,
                                                float* __restrict__ out_emb) {
    int gid = blockIdx.x * 256 + threadIdx.x;     // NCODES*64
    int k = gid >> 6;
    float n = 0.f;
#pragma unroll
    for (int i = 0; i < 16; ++i) n += npart[i];
    float cs = (out_ncs[k] + EPS) / (n + (float)NCODES * EPS) * n;
    out_emb[gid] = out_nemaw[gid] / cs;
}

// -------------------- gather z_q + per-block loss partials --------------------
__global__ __launch_bounds__(256) void k_quant(const float* __restrict__ z,
                                               const int* __restrict__ codes_i,
                                               const float* __restrict__ out_emb,
                                               float* __restrict__ zq,
                                               float* __restrict__ lpart) {
    int gid = blockIdx.x * 256 + threadIdx.x;     // NROWS*64
    int row = gid >> 6, d = gid & 63;
    int c = codes_i[row];
    float q = out_emb[(size_t)c * DIM + d];
    float x = z[gid];
    zq[gid] = q;                                  // z_q_st == z_q in value
    float diff = q - x;
    __shared__ float sm[256];
    sm[threadIdx.x] = diff * diff;
    __syncthreads();
    for (int s = 128; s > 0; s >>= 1) {
        if (threadIdx.x < s) sm[threadIdx.x] += sm[threadIdx.x + s];
        __syncthreads();
    }
    if (threadIdx.x == 0) lpart[blockIdx.x] = sm[0];
}

// -------------------- final deterministic loss reduction --------------------
__global__ __launch_bounds__(256) void k_loss(const float* __restrict__ lpart, float* __restrict__ out) {
    __shared__ float sm[256];
    float s = 0.f;
    for (int i = threadIdx.x; i < ZELEMS / 256; i += 256) s += lpart[i];  // 18432 partials
    sm[threadIdx.x] = s;
    __syncthreads();
    for (int k = 128; k > 0; k >>= 1) {
        if (threadIdx.x < k) sm[threadIdx.x] += sm[threadIdx.x + k];
        __syncthreads();
    }
    if (threadIdx.x == 0) out[OFF_LOSS] = CCOST * sm[0] / (float)ZELEMS;
}

extern "C" void kernel_launch(void* const* d_in, const int* in_sizes, int n_in,
                              void* d_out, int out_size, void* d_ws, size_t ws_size,
                              hipStream_t stream) {
    const float* z    = (const float*)d_in[0];   // [128,24,1536]
    const float* emb  = (const float*)d_in[1];   // [4096,64]
    const float* ecs  = (const float*)d_in[2];   // [4096]
    const float* emaw = (const float*)d_in[3];   // [4096,64]
    float* out = (float*)d_out;
    float*    wsf = (float*)d_ws;
    uint32_t* wsu = (uint32_t*)d_ws;
    int*      wsi = (int*)d_ws;

    uint32_t* bhi    = wsu + WS_BHI;
    uint32_t* blo    = wsu + WS_BLO;
    float*    enorm  = wsf + WS_ENORM;
    float*    counts = wsf + WS_CNT;
    float*    dw     = wsf + WS_DW;
    int*      codes  = wsi + WS_CODE;
    float*    npart  = wsf + WS_NPART;
    float*    lpart  = wsf + WS_LPART;

    k_zero<<<(ZERO_CNT + 255) / 256, 256, 0, stream>>>(wsf);
    k_enorm<<<NCODES / 256, 256, 0, stream>>>(emb, enorm);
    k_packB<<<(NKT * 2 * 32) / 256, 256, 0, stream>>>(emb, bhi, blo);
    k_argmin<<<NROWS / 128, 128, 0, stream>>>(z, bhi, blo, enorm, codes, out + OFF_CODE);
    k_scatter<<<ZELEMS / 256, 256, 0, stream>>>(z, codes, counts, dw);
    k_ema<<<NCODES / 256, 256, 0, stream>>>(ecs, emaw, counts, dw,
                                            out + OFF_NCS, out + OFF_NEW, npart);
    k_newemb<<<(NCODES * DIM) / 256, 256, 0, stream>>>(out + OFF_NCS, out + OFF_NEW, npart,
                                                       out + OFF_EMB);
    k_quant<<<ZELEMS / 256, 256, 0, stream>>>(z, codes, out + OFF_EMB, out + OFF_ZQ, lpart);
    k_loss<<<1, 256, 0, stream>>>(lpart, out);
}
